// PatchInpainting_53188874993772
// MI455X (gfx1250) — matrix operand order repaired and checked by reading, not compiled, not run on hardware
//
#include <hip/hip_runtime.h>

// ---------------------------------------------------------------------------
// Problem constants (from the reference)
// ---------------------------------------------------------------------------
#define BB     4
#define LL     1024
#define DTOK   897
#define NHEAD  8
#define DQK    512
#define DV     768
#define TOPK   8
#define NEGV   (-10000.0f)

typedef __attribute__((ext_vector_type(16))) __bf16 v16bf;
typedef __attribute__((ext_vector_type(8)))  float  v8f;

union FragB16 { v16bf v; unsigned int u[8]; };
union FragAcc { v8f   v; float        f[8]; };

static __device__ __forceinline__ unsigned short f32_to_bf16(float f) {
    unsigned int u = __float_as_uint(f);
    unsigned int rnd = 0x7FFFu + ((u >> 16) & 1u);   // round to nearest even
    u += rnd;
    return (unsigned short)(u >> 16);
}
static __device__ __forceinline__ unsigned int pack_bf16x2(float lo, float hi) {
    return (unsigned int)f32_to_bf16(lo) | ((unsigned int)f32_to_bf16(hi) << 16);
}

// ---------------------------------------------------------------------------
// Tiled bf16-WMMA GEMM.  Block = 256 threads (8 wave32), tile 64(M)x128(N),
// K step 32.  Each wave owns a 16x64 strip -> 4 accumulators, A frag reused.
// Main K loop is guard-free & vectorized; only the K%32 tail (K=897 proj)
// takes a guarded scalar path.
//
// STAGE 0: projection.  A=f32 [M,K], B=f32 [K,N] -> bf16 [B,H,L,Dh] (+scale).
// STAGE 1: attn scores. A=bf16 qh [L,DQK], B=bf16 kh [N,K], batched z=b*8+h.
//          Epilogue adds -1e4*eye + k_mask, f32 out.
// STAGE 2: attn @ V.    A=f32 probs [L,L], B=bf16 vh [K,N], batched ->
//          bf16 head-concat [B*L, H*DV].
// STAGE 3: final fc.    A=bf16 concat, B=f32 w_fc -> plain f32 out.
// ---------------------------------------------------------------------------
template<int STAGE>
__global__ __launch_bounds__(256)
void gemm_wmma_kernel(const void* __restrict__ Ap,
                      const void* __restrict__ Bp,
                      void*       __restrict__ Cp,
                      const float* __restrict__ kmask,
                      int M, int N, int K, int lda, int ldb,
                      long strideA, long strideB, long strideC,
                      float scale, int Dh)
{
    constexpr bool A_IS_F32 = (STAGE == 0 || STAGE == 2);
    constexpr bool B_IS_F32 = (STAGE == 0 || STAGE == 3);
    constexpr bool B_NK     = (STAGE == 1);   // B given as row-major [N,K]

    __shared__ unsigned short ldsA[64][34];   // padded: 68B rows, 4B aligned
    __shared__ unsigned short ldsB[128][34];

    const int tid   = threadIdx.x;
    const int wave  = tid >> 5;
    const int lane  = tid & 31;
    const int half  = lane >> 4;
    const int laneN = lane & 15;
    const int waveM = wave & 3;               // 4 waves along M
    const int waveN = wave >> 2;              // 2 waves along N
    const int z     = blockIdx.z;
    const int blockM = blockIdx.y * 64;
    const int blockN = blockIdx.x * 128;

    const float*          Af = (const float*)Ap          + (A_IS_F32 ? (long)z * strideA : 0);
    const unsigned short* Ah = (const unsigned short*)Ap + (A_IS_F32 ? 0 : (long)z * strideA);
    const float*          Bf = (const float*)Bp          + (B_IS_F32 ? (long)z * strideB : 0);
    const unsigned short* Bh = (const unsigned short*)Bp + (B_IS_F32 ? 0 : (long)z * strideB);

    FragAcc acc[4];
#pragma unroll
    for (int t = 0; t < 4; ++t)
#pragma unroll
        for (int r = 0; r < 8; ++r) acc[t].f[r] = 0.0f;

    // one WMMA step on the currently staged LDS tiles
    auto mma_step = [&]() {
        const int mrow = waveM * 16 + laneN;
        FragB16 a;
#pragma unroll
        for (int v = 0; v < 4; ++v)
            a.u[v]     = *(const unsigned int*)&ldsA[mrow][half * 16 + v * 2];
#pragma unroll
        for (int v = 0; v < 4; ++v)
            a.u[4 + v] = *(const unsigned int*)&ldsA[mrow][half * 16 + 8 + v * 2];
#pragma unroll
        for (int t = 0; t < 4; ++t) {
            const int nrow = waveN * 64 + t * 16 + laneN;
            FragB16 b;
#pragma unroll
            for (int v = 0; v < 8; ++v)
                b.u[v] = *(const unsigned int*)&ldsB[nrow][half * 16 + v * 2];
            acc[t].v = __builtin_amdgcn_wmma_f32_16x16x32_bf16(
                false, a.v, false, b.v, (short)0, acc[t].v, false, false);
        }
    };

    const int Kfull = K & ~31;
    for (int k0 = 0; k0 < Kfull; k0 += 32) {
        // ---- A tile 64x32: thread -> (row = t/4, 8 cols), branch-free ----
        {
            const int row = tid >> 2, cb = (tid & 3) * 8;
            unsigned int* dst = (unsigned int*)&ldsA[row][cb];
            if (A_IS_F32) {
                const float* src = Af + (long)(blockM + row) * lda + (k0 + cb);
                float f[8];
                if ((lda & 3) == 0) {               // 16B-aligned rows
                    float4 x = *(const float4*)src;
                    float4 y = *(const float4*)(src + 4);
                    f[0]=x.x; f[1]=x.y; f[2]=x.z; f[3]=x.w;
                    f[4]=y.x; f[5]=y.y; f[6]=y.z; f[7]=y.w;
                } else {                            // lda=897: dword loads
#pragma unroll
                    for (int j = 0; j < 8; ++j) f[j] = src[j];
                }
#pragma unroll
                for (int j = 0; j < 4; ++j) dst[j] = pack_bf16x2(f[2*j], f[2*j+1]);
            } else {
                uint4 x = *(const uint4*)(Ah + (long)(blockM + row) * lda + (k0 + cb));
                dst[0] = x.x; dst[1] = x.y; dst[2] = x.z; dst[3] = x.w;
            }
        }
        // ---- B tile 128x32 -> LDS [N][K] ----
        if (B_NK) {           // bf16 [N,K]: straight contiguous copy
            const int n = tid >> 1, cb = (tid & 1) * 16;
            const uint4* src = (const uint4*)(Bh + (long)(blockN + n) * ldb + (k0 + cb));
            uint4 x = src[0], y = src[1];
            unsigned int* dst = (unsigned int*)&ldsB[n][cb];
            dst[0]=x.x; dst[1]=x.y; dst[2]=x.z; dst[3]=x.w;
            dst[4]=y.x; dst[5]=y.y; dst[6]=y.z; dst[7]=y.w;
        } else if (B_IS_F32) { // f32 [K,N]: wide row loads, transposed stores
#pragma unroll
            for (int i = 0; i < 2; ++i) {
                const int kk = (tid >> 4) + i * 16;
                const int nb = (tid & 15) * 8;
                const float* src = Bf + (long)(k0 + kk) * ldb + blockN + nb;
                float4 x = *(const float4*)src;
                float4 y = *(const float4*)(src + 4);
                float f[8] = {x.x,x.y,x.z,x.w,y.x,y.y,y.z,y.w};
#pragma unroll
                for (int j = 0; j < 8; ++j) ldsB[nb + j][kk] = f32_to_bf16(f[j]);
            }
        } else {               // bf16 [K,N] (vh): wide row loads, transposed
#pragma unroll
            for (int i = 0; i < 2; ++i) {
                const int kk = (tid >> 4) + i * 16;
                const int nb = (tid & 15) * 8;
                uint4 x = *(const uint4*)(Bh + (long)(k0 + kk) * ldb + blockN + nb);
                unsigned int w[4] = {x.x, x.y, x.z, x.w};
#pragma unroll
                for (int j = 0; j < 4; ++j) {
                    ldsB[nb + 2*j    ][kk] = (unsigned short)(w[j] & 0xFFFFu);
                    ldsB[nb + 2*j + 1][kk] = (unsigned short)(w[j] >> 16);
                }
            }
        }
        // prefetch next K tile (gfx1250 global_prefetch path)
        if (k0 + 32 < Kfull) {
            if (A_IS_F32) __builtin_prefetch(&Af[(long)(blockM + (tid >> 2)) * lda + k0 + 32], 0, 0);
            else          __builtin_prefetch(&Ah[(long)(blockM + (tid >> 2)) * lda + k0 + 32], 0, 0);
        }
        __syncthreads();
        mma_step();
        __syncthreads();
    }

    // ---- guarded tail tile (only for K % 32 != 0, i.e. K=897 proj) ----
    if (Kfull < K) {
        const int k0 = Kfull;
#pragma unroll
        for (int i = 0; i < 8; ++i) {
            int idx = i * 256 + tid;
            int r = idx >> 5, c = idx & 31;
            int gm = blockM + r, gk = k0 + c;
            unsigned short us = 0;
            if (gm < M && gk < K)
                us = A_IS_F32 ? f32_to_bf16(Af[(long)gm * lda + gk])
                              : Ah[(long)gm * lda + gk];
            ldsA[r][c] = us;
        }
#pragma unroll
        for (int i = 0; i < 16; ++i) {
            int idx = i * 256 + tid;
            int n, c;
            if (B_NK) { n = idx >> 5;  c = idx & 31; }
            else      { c = idx >> 7;  n = idx & 127; }
            int gn = blockN + n, gk = k0 + c;
            unsigned short us = 0;
            if (gn < N && gk < K) {
                long o = B_NK ? ((long)gn * ldb + gk) : ((long)gk * ldb + gn);
                us = B_IS_F32 ? f32_to_bf16(Bf[o]) : Bh[o];
            }
            ldsB[n][c] = us;
        }
        __syncthreads();
        mma_step();
        __syncthreads();
    }

    // ---- epilogue: C/D layout  M = r + 8*half, N = lane&15 ----
#pragma unroll
    for (int t = 0; t < 4; ++t) {
#pragma unroll
        for (int r = 0; r < 8; ++r) {
            int gm = blockM + waveM * 16 + half * 8 + r;
            int gn = blockN + waveN * 64 + t * 16 + laneN;
            if (gm >= M || gn >= N) continue;
            float c = acc[t].f[r];
            if (STAGE == 0) {
                int b = gm >> 10, l = gm & 1023;
                int h = gn / Dh,  d = gn % Dh;
                unsigned short* Co = (unsigned short*)Cp;
                Co[((long)((b * NHEAD + h) * LL + l)) * Dh + d] = f32_to_bf16(c * scale);
            } else if (STAGE == 1) {
                int b = z >> 3;
                c += (gm == gn) ? NEGV : 0.0f;
                c += kmask[b * LL + gn];
                float* Co = (float*)Cp + (long)z * strideC;
                Co[(long)gm * N + gn] = c;
            } else if (STAGE == 2) {
                int b = z >> 3, h = z & 7;
                unsigned short* Co = (unsigned short*)Cp;
                Co[((long)(b * LL + gm)) * (NHEAD * DV) + h * DV + gn] = f32_to_bf16(c);
            } else {
                float* Co = (float*)Cp;
                Co[(long)gm * N + gn] = c;
            }
        }
    }
}

// ---------------------------------------------------------------------------
// Top-k(8) sparsify + softmax, one block per attention row (in place, f32).
// Non-top-k entries get prob 0 (exp(-1e4 - m) underflows to 0 in the ref).
// ---------------------------------------------------------------------------
__global__ __launch_bounds__(256)
void topk_softmax_kernel(float* __restrict__ attn)
{
    const long row = blockIdx.x;
    float* p = attn + row * (long)LL;

    __shared__ float vals[LL];
    __shared__ float redV[256];
    __shared__ int   redI[256];
    __shared__ float topv[TOPK];
    __shared__ int   topi[TOPK];

    const int t = threadIdx.x;
#pragma unroll
    for (int i = 0; i < 4; ++i) vals[t + i * 256] = p[t + i * 256];
    __syncthreads();

    for (int r = 0; r < TOPK; ++r) {
        float bv = -__builtin_inff(); int bi = 0x7FFFFFFF;
#pragma unroll
        for (int i = 0; i < 4; ++i) {
            int j = t + i * 256; float v = vals[j];
            if (v > bv || (v == bv && j < bi)) { bv = v; bi = j; }
        }
        redV[t] = bv; redI[t] = bi; __syncthreads();
        for (int s = 128; s > 0; s >>= 1) {
            if (t < s) {
                float v2 = redV[t + s]; int i2 = redI[t + s];
                if (v2 > redV[t] || (v2 == redV[t] && i2 < redI[t])) {
                    redV[t] = v2; redI[t] = i2;
                }
            }
            __syncthreads();
        }
        if (t == 0) {
            topv[r] = redV[0]; topi[r] = redI[0];
            vals[redI[0]] = -__builtin_inff();
        }
        __syncthreads();
    }

    // zero the row, then scatter the 8 softmax probabilities
#pragma unroll
    for (int i = 0; i < 4; ++i) p[t + i * 256] = 0.0f;
    __syncthreads();
    if (t < TOPK) {
        float m = topv[0];
        float s = 0.0f;
#pragma unroll
        for (int r = 0; r < TOPK; ++r) s += __expf(topv[r] - m);
        p[topi[t]] = __expf(topv[t] - m) / s;
    }
}

// ---------------------------------------------------------------------------
// Host-side orchestration
// ---------------------------------------------------------------------------
extern "C" void kernel_launch(void* const* d_in, const int* in_sizes, int n_in,
                              void* d_out, int out_size, void* d_ws, size_t ws_size,
                              hipStream_t stream)
{
    (void)in_sizes; (void)n_in; (void)out_size; (void)ws_size;

    const float* q     = (const float*)d_in[0];
    const float* k     = (const float*)d_in[1];
    const float* v     = (const float*)d_in[2];
    /* d_in[3] = qk_mask (-1e4*eye): applied analytically in stage 1 */
    const float* kmask = (const float*)d_in[4];
    const float* w_qs  = (const float*)d_in[5];
    const float* w_ks  = (const float*)d_in[6];
    const float* w_vs  = (const float*)d_in[7];
    const float* w_fc  = (const float*)d_in[8];

    // workspace carve (bytes):
    //   vh   bf16 [B,H,L,DV] : 50,331,648
    //   qh   bf16 [B,H,L,DQK]: 33,554,432  (dead after stage 1)
    //   kh   bf16 [B,H,L,DQK]: 33,554,432  (dead after stage 1)
    //   concat bf16 [B*L, H*DV]: 50,331,648 (overlaps qh+kh region)
    char* ws = (char*)d_ws;
    unsigned short* vh     = (unsigned short*)ws;
    unsigned short* qh     = (unsigned short*)(ws + 50331648L);
    unsigned short* kh     = (unsigned short*)(ws + 50331648L + 33554432L);
    unsigned short* concat = (unsigned short*)(ws + 50331648L);

    float* out_buf  = (float*)d_out;                       // [B,L,DV]
    float* attn_buf = out_buf + (long)BB * LL * DV;        // [B,H,L,L]

    const int ML = BB * LL;            // 4096
    const float qscale = 0.044194173824159216f;  // 1/sqrt(512)

    // 1) qh = (q @ w_qs) * scale  -> bf16 [B,H,L,DQK]
    gemm_wmma_kernel<0><<<dim3((NHEAD * DQK) / 128, ML / 64, 1), 256, 0, stream>>>(
        q, w_qs, qh, nullptr, ML, NHEAD * DQK, DTOK, DTOK, NHEAD * DQK,
        0, 0, 0, qscale, DQK);
    // 2) kh = k @ w_ks -> bf16 [B,H,L,DQK]
    gemm_wmma_kernel<0><<<dim3((NHEAD * DQK) / 128, ML / 64, 1), 256, 0, stream>>>(
        k, w_ks, kh, nullptr, ML, NHEAD * DQK, DTOK, DTOK, NHEAD * DQK,
        0, 0, 0, 1.0f, DQK);
    // 3) vh = v @ w_vs -> bf16 [B,H,L,DV]
    gemm_wmma_kernel<0><<<dim3((NHEAD * DV) / 128, ML / 64, 1), 256, 0, stream>>>(
        v, w_vs, vh, nullptr, ML, NHEAD * DV, DV, DV, NHEAD * DV,
        0, 0, 0, 1.0f, DV);
    // 4) scores = qh @ kh^T + masks -> f32 attn_buf, batched over 32 heads
    gemm_wmma_kernel<1><<<dim3(LL / 128, LL / 64, BB * NHEAD), 256, 0, stream>>>(
        qh, kh, attn_buf, kmask, LL, LL, DQK, DQK, DQK,
        (long)LL * DQK, (long)LL * DQK, (long)LL * LL, 1.0f, 0);
    // 5) top-k sparsify + softmax in place
    topk_softmax_kernel<<<BB * NHEAD * LL, 256, 0, stream>>>(attn_buf);
    // 6) heads = attn @ vh -> bf16 concat [B*L, H*DV], batched
    gemm_wmma_kernel<2><<<dim3(DV / 128, LL / 64, BB * NHEAD), 256, 0, stream>>>(
        attn_buf, vh, concat, nullptr, LL, DV, LL, LL, DV,
        (long)LL * LL, (long)LL * DV, 0, 1.0f, 0);
    // 7) out = concat @ w_fc -> f32 [B*L, DV]
    gemm_wmma_kernel<3><<<dim3(DV / 128, ML / 64, 1), 256, 0, stream>>>(
        concat, w_fc, out_buf, nullptr, ML, DV, NHEAD * DV, NHEAD * DV, DV,
        0, 0, 0, 1.0f, 0);
}
